// CTSPd_Decoder_39986145525835
// MI455X (gfx1250) — matrix operand order repaired and checked by reading, not compiled, not run on hardware
//
#include <hip/hip_runtime.h>
#include <hip/hip_bf16.h>
#include <math.h>

// Problem constants (match reference)
#define BB   64
#define PP   100
#define NN   1000
#define DD   128
#define HH   8
#define QDQ  16
#define NPAD 1008   // 63 * 16 (n tiles)
#define KPAD 1024   // padded N for PV contraction (32 WMMA chunks of 32)

typedef __bf16 bf16_t;
typedef __attribute__((ext_vector_type(16))) __bf16 v16bf;
typedef __attribute__((ext_vector_type(8)))  __bf16 v8bf;
typedef __attribute__((ext_vector_type(4)))  __bf16 bf16x4;
typedef __attribute__((ext_vector_type(8)))  float  v8f;
typedef __attribute__((ext_vector_type(4)))  float  f32x4;

// ---------- helpers ----------
__device__ inline bf16_t f2bf(float f) { return (bf16_t)f; }  // native v_cvt path

__device__ inline v8f wmma_bf16(v16bf a, v16bf b, v8f c) {
    // D = A(16x32 bf16) * B(32x16 bf16) + C(16x16 f32)
    return __builtin_amdgcn_wmma_f32_16x16x32_bf16(
        /*neg_a=*/false, a, /*neg_b=*/false, b,
        /*c_mod=*/(short)0, c, /*reuse_a=*/false, /*reuse_b=*/false);
}

// CDNA5 async copy: 16B global -> LDS per lane, tracked on ASYNCcnt.
__device__ inline void async_copy_b128(void* lds_dst, const void* gsrc) {
    unsigned loff = (unsigned)(size_t)lds_dst;  // low 32 bits of generic ptr = LDS offset
    asm volatile("global_load_async_to_lds_b128 %0, %1, off"
                 :: "v"(loff), "v"(gsrc) : "memory");
}
__device__ inline void wait_async0() {
    asm volatile("s_wait_asynccnt 0x0" ::: "memory");
}

// A-matrix (16x32, 16-bit) per-lane layout:
//  lanes 0-15 : e0-7 -> K kb+0..7,  e8-15 -> K kb+8..15
//  lanes 16-31: e0-7 -> K kb+16..23, e8-15 -> K kb+24..31
// => two contiguous 8-element groups at (kb + half*16) and (kb + half*16 + 8); this code's order, same as B.
// B-matrix (32x16, 16-bit) per-lane layout: col = lane&15, elements = 16
// contiguous K values at (kb + half*16).

__device__ inline v16bf cvt16(f32x4 u0, f32x4 u1, f32x4 u2, f32x4 u3) {
    bf16x4 c0 = __builtin_convertvector(u0, bf16x4);
    bf16x4 c1 = __builtin_convertvector(u1, bf16x4);
    bf16x4 c2 = __builtin_convertvector(u2, bf16x4);
    bf16x4 c3 = __builtin_convertvector(u3, bf16x4);
    v16bf a;
#pragma unroll
    for (int e = 0; e < 4; ++e) {
        a[e] = c0[e]; a[4 + e] = c1[e]; a[8 + e] = c2[e]; a[12 + e] = c3[e];
    }
    return a;
}

// Load A fragment from an f32 row.
__device__ inline v16bf load_a_f32(const float* __restrict__ row, int kb, int half, bool valid) {
    const float* g0 = row + kb + half * 16;
    const float* g1 = row + kb + half * 16 + 8;
    v16bf a = cvt16(*(const f32x4*)(g0), *(const f32x4*)(g0 + 4),
                    *(const f32x4*)(g1), *(const f32x4*)(g1 + 4));
    v16bf z = {};
    return valid ? a : z;
}

// Load A fragment from a bf16 row (global or LDS).
__device__ inline v16bf load_a_bf16(const bf16_t* __restrict__ row, int kb, int half, bool valid) {
    v8bf g0 = *(const v8bf*)(row + kb + half * 16);
    v8bf g1 = *(const v8bf*)(row + kb + half * 16 + 8);
    v16bf a;
#pragma unroll
    for (int e = 0; e < 8; ++e) { a[e] = g0[e]; a[8 + e] = g1[e]; }
    v16bf z = {};
    return valid ? a : z;
}

// Load B fragment: 16 contiguous f32 -> bf16.
__device__ inline v16bf load_b_f32(const float* __restrict__ base, bool valid) {
    v16bf b = cvt16(((const f32x4*)base)[0], ((const f32x4*)base)[1],
                    ((const f32x4*)base)[2], ((const f32x4*)base)[3]);
    v16bf z = {};
    return valid ? b : z;
}

__device__ inline float wave_max(float v) {
    for (int o = 16; o > 0; o >>= 1) v = fmaxf(v, __shfl_xor(v, o));
    return v;
}
__device__ inline float wave_sum(float v) {
    for (int o = 16; o > 0; o >>= 1) v += __shfl_xor(v, o);
    return v;
}

// ---------- Kprep: transpose 5 weight matrices f32[kin][nout] -> bf16 WT[m][nout][kin] ----------
__global__ void kprep_wt(const float* __restrict__ W0, const float* __restrict__ W1,
                         const float* __restrict__ W2, const float* __restrict__ W3,
                         const float* __restrict__ W4, bf16_t* __restrict__ WT) {
    int nout = blockIdx.x;
    int m = blockIdx.y;
    int k = threadIdx.x;
    const float* W = (m == 0) ? W0 : (m == 1) ? W1 : (m == 2) ? W2 : (m == 3) ? W3 : W4;
    WT[((size_t)m * DD + nout) * DD + k] = f2bf(W[(size_t)k * DD + nout]);
}

// ---------- K0: q = heads(q1 @ Wq_first + last @ Wq_last), bf16 out [B,P,128] ----------
__global__ void k0_proj_q(const float* __restrict__ eq1, const float* __restrict__ elast,
                          const bf16_t* __restrict__ WqfT, const bf16_t* __restrict__ WqlT,
                          bf16_t* __restrict__ q_bf) {
    int pt = blockIdx.x, b = blockIdx.y;
    int lane = threadIdx.x & 31, half = lane >> 4, l15 = lane & 15;
    int prow = pt * 16 + l15;
    bool pv = prow < PP;
    const float* r1 = eq1 + ((size_t)b * PP + (pv ? prow : 0)) * DD;
    const float* r2 = elast + ((size_t)b * PP + (pv ? prow : 0)) * DD;
    v16bf aa1[4], aa2[4];
#pragma unroll
    for (int kc = 0; kc < 4; ++kc) {
        aa1[kc] = load_a_f32(r1, kc * 32, half, pv);
        aa2[kc] = load_a_f32(r2, kc * 32, half, pv);
    }
    for (int ct = 0; ct < 8; ++ct) {
        const bf16_t* w1 = WqfT + (size_t)(ct * 16 + l15) * DD + half * 16;
        const bf16_t* w2 = WqlT + (size_t)(ct * 16 + l15) * DD + half * 16;
        v16bf b1[4], b2[4];
#pragma unroll
        for (int kc = 0; kc < 4; ++kc) {  // batch all B loads: one clause, one wait
            b1[kc] = *(const v16bf*)(w1 + kc * 32);
            b2[kc] = *(const v16bf*)(w2 + kc * 32);
        }
        v8f c = {};
#pragma unroll
        for (int kc = 0; kc < 4; ++kc) {
            c = wmma_bf16(aa1[kc], b1[kc], c);
            c = wmma_bf16(aa2[kc], b2[kc], c);
        }
#pragma unroll
        for (int r = 0; r < 8; ++r) {
            int p = pt * 16 + r + half * 8;
            if (p < PP) q_bf[((size_t)b * PP + p) * DD + ct * 16 + l15] = f2bf(c[r]);
        }
    }
}

// ---------- K1: k = nodes@Wk (bf16 [B,N,128]), v = nodes@Wv transposed (bf16 [B,128,1024]) ----------
__global__ void k1_proj_kv(const float* __restrict__ nodes,
                           const bf16_t* __restrict__ WkT, const bf16_t* __restrict__ WvT,
                           bf16_t* __restrict__ k_bf, bf16_t* __restrict__ vT) {
    int nt = blockIdx.x, b = blockIdx.y;
    int lane = threadIdx.x & 31, half = lane >> 4, l15 = lane & 15;
    int nrow = nt * 16 + l15;
    bool nv = nrow < NN;
    const float* ar = nodes + ((size_t)b * NN + (nv ? nrow : 0)) * DD;
    v16bf aa[4];
#pragma unroll
    for (int kc = 0; kc < 4; ++kc) aa[kc] = load_a_f32(ar, kc * 32, half, nv);
    for (int ct = 0; ct < 8; ++ct) {
        const bf16_t* wk = WkT + (size_t)(ct * 16 + l15) * DD + half * 16;
        const bf16_t* wv = WvT + (size_t)(ct * 16 + l15) * DD + half * 16;
        v16bf bk[4], bv[4];
#pragma unroll
        for (int kc = 0; kc < 4; ++kc) {
            bk[kc] = *(const v16bf*)(wk + kc * 32);
            bv[kc] = *(const v16bf*)(wv + kc * 32);
        }
        v8f ck = {}, cv = {};
#pragma unroll
        for (int kc = 0; kc < 4; ++kc) {
            ck = wmma_bf16(aa[kc], bk[kc], ck);
            cv = wmma_bf16(aa[kc], bv[kc], cv);
        }
        // k: scalar strided stores (row-major [B,N,128])
#pragma unroll
        for (int r = 0; r < 8; ++r) {
            int n = nt * 16 + r + half * 8;
            if (n < NN) k_bf[((size_t)b * NN + n) * DD + ct * 16 + l15] = f2bf(ck[r]);
        }
        // v^T: 8 contiguous bf16 per lane along n
        v8bf vv;
#pragma unroll
        for (int r = 0; r < 8; ++r) vv[r] = f2bf(cv[r]);
        int n0 = nt * 16 + half * 8;
        bf16_t* vtp = vT + ((size_t)b * DD + ct * 16 + l15) * KPAD + n0;
        if (n0 + 8 <= NN) {
            *(v8bf*)vtp = vv;
        } else {
#pragma unroll
            for (int r = 0; r < 8; ++r)
                if (n0 + r < NN) vtp[r] = vv[r];
        }
    }
}

// ---------- K2: per (b,h,ptile): QK^T, masked softmax (mask async-prefetched to LDS), PV ----------
__global__ void k2_attention(const bf16_t* __restrict__ q_bf, const bf16_t* __restrict__ k_bf,
                             const bf16_t* __restrict__ vT, const float* __restrict__ mask,
                             bf16_t* __restrict__ outc) {
    extern __shared__ char smem[];
    float*  sc   = (float*)smem;                                    // 16 x NPAD  f32  (64512)
    bf16_t* wts  = (bf16_t*)(smem + 16 * NPAD * 4);                 // 16 x KPAD  bf16 (32768)
    float*  acc  = (float*)(smem + 16 * NPAD * 4 + 16 * KPAD * 2);  // 16 x 16    f32  (1024)
    float*  mlds = (float*)(smem + 16 * NPAD * 4 + 16 * KPAD * 2 + 1024);  // 16x1000 f32 (64000)
    int pt = blockIdx.x, h = blockIdx.y, b = blockIdx.z;
    int tid = threadIdx.x, wid = tid >> 5, lane = tid & 31;
    int half = lane >> 4, l15 = lane & 15;
    acc[tid] = 0.f;

    // ---- kick off async mask panel copy (16 rows x 1000 f32 = 4000 x 16B chunks) ----
    for (int cch = tid; cch < 4000; cch += 256) {
        int row = cch / 250;
        int off = cch - row * 250;  // 16B chunk within row
        int p = pt * 16 + row;
        const float* g = mask + ((size_t)b * PP + (p < PP ? p : 0)) * NN + off * 4;
        async_copy_b128(mlds + row * 1000 + off * 4, g);
    }

    // ---- phase 1: scores = q . k  (per-head contraction over 16 dims) ----
    int prow = pt * 16 + l15;
    bool pvld = prow < PP;
    const bf16_t* qrow = q_bf + ((size_t)b * PP + (pvld ? prow : 0)) * DD + h * QDQ;
    v16bf zv = {};
    v16bf aq;
    {
        v8bf g0 = *(const v8bf*)(qrow);  // half 0: K = 0..15 contiguous; half 1 (K>=16) is zero
        v8bf g1 = *(const v8bf*)(qrow + 8);
#pragma unroll
        for (int e = 0; e < 8; ++e) { aq[e] = g0[e]; aq[8 + e] = g1[e]; }
        aq = (pvld && half == 0) ? aq : zv;
    }
    for (int t = wid; t < 63; t += 8) {
        int nb = t * 16;
        int n = nb + l15;
        const bf16_t* krow = k_bf + ((size_t)b * NN + (n < NN ? n : 0)) * DD + h * QDQ;
        if (t + 8 < 63) {
            __builtin_prefetch(k_bf + ((size_t)b * NN + (t + 8) * 16 + l15) * DD + h * QDQ, 0, 1);
        }
        // B fragment: half 0 lanes carry K=0..15 (the real head dims), half 1 lanes are zero.
        v16bf kraw = *(const v16bf*)(krow + half * 16);
        v16bf bk = (half == 0 && n < NN) ? kraw : zv;
        v8f c = {};
        c = wmma_bf16(aq, bk, c);
        if (n < NN) {
#pragma unroll
            for (int r = 0; r < 8; ++r) sc[(r + 8 * half) * NPAD + n] = c[r];
        }
    }
    wait_async0();   // our async chunks done; barrier makes all waves' chunks visible
    __syncthreads();

    // ---- phase 2: masked softmax rows -> normalized bf16 weights in LDS ----
    for (int rr = 0; rr < 2; ++rr) {
        int row = wid * 2 + rr;
        int p = pt * 16 + row;
        if (p >= PP) {  // wave-uniform branch
            bf16_t bz = f2bf(0.f);
            for (int n = lane; n < KPAD; n += 32) wts[row * KPAD + n] = bz;
            continue;
        }
        const float* mrow = mlds + row * 1000;
        float mx = -3.0e38f;
        for (int n = lane; n < NN; n += 32) {
            float s = sc[row * NPAD + n] * 0.25f + mrow[n];  // 1/sqrt(QD) = 0.25
            sc[row * NPAD + n] = s;
            mx = fmaxf(mx, s);
        }
        mx = wave_max(mx);
        float sum = 0.f;
        for (int n = lane; n < NN; n += 32) {
            float e = __expf(sc[row * NPAD + n] - mx);
            sc[row * NPAD + n] = e;
            sum += e;
        }
        sum = wave_sum(sum);
        float inv = 1.0f / sum;
        for (int n = lane; n < NN; n += 32) wts[row * KPAD + n] = f2bf(sc[row * NPAD + n] * inv);
        bf16_t bz = f2bf(0.f);
        for (int n = NN + lane; n < KPAD; n += 32) wts[row * KPAD + n] = bz;
    }
    __syncthreads();

    // ---- phase 3: out = weights @ v  (K=1024, 4 chunks per wave) ----
    const bf16_t* wrow = wts + (size_t)l15 * KPAD;
    const bf16_t* vrow = vT + ((size_t)b * DD + h * QDQ + l15) * KPAD;
    v16bf aw[4], bv4[4];
#pragma unroll
    for (int i = 0; i < 4; ++i) {
        int kb = (wid * 4 + i) * 32;
        aw[i]  = load_a_bf16(wrow, kb, half, true);       // two ds_load_b128
        bv4[i] = *(const v16bf*)(vrow + kb + half * 16);  // contiguous 32B
    }
    v8f co = {};
#pragma unroll
    for (int i = 0; i < 4; ++i) co = wmma_bf16(aw[i], bv4[i], co);
#pragma unroll
    for (int r = 0; r < 8; ++r)
        atomicAdd(&acc[(r + 8 * half) * 16 + l15], co[r]);
    __syncthreads();

    int row = tid >> 4, col = tid & 15;
    int p = pt * 16 + row;
    if (p < PP) outc[((size_t)b * PP + p) * DD + h * QDQ + col] = f2bf(acc[row * 16 + col]);
}

// ---------- K3: mh = out_concat @ W_comb + b_comb, bf16 out [B,P,128] ----------
__global__ void k3_comb(const bf16_t* __restrict__ outc, const bf16_t* __restrict__ WcT,
                        const float* __restrict__ bias, bf16_t* __restrict__ mh) {
    int pt = blockIdx.x, b = blockIdx.y;
    int lane = threadIdx.x & 31, half = lane >> 4, l15 = lane & 15;
    int prow = pt * 16 + l15;
    bool pv = prow < PP;
    const bf16_t* ar = outc + ((size_t)b * PP + (pv ? prow : 0)) * DD;
    v16bf aa[4];
#pragma unroll
    for (int kc = 0; kc < 4; ++kc) aa[kc] = load_a_bf16(ar, kc * 32, half, pv);
    for (int ct = 0; ct < 8; ++ct) {
        const bf16_t* wc = WcT + (size_t)(ct * 16 + l15) * DD + half * 16;
        v16bf bw[4];
#pragma unroll
        for (int kc = 0; kc < 4; ++kc) bw[kc] = *(const v16bf*)(wc + kc * 32);
        v8f c = {};
#pragma unroll
        for (int kc = 0; kc < 4; ++kc) c = wmma_bf16(aa[kc], bw[kc], c);
        float bb = bias[ct * 16 + l15];
#pragma unroll
        for (int r = 0; r < 8; ++r) {
            int p = pt * 16 + r + half * 8;
            if (p < PP) mh[((size_t)b * PP + p) * DD + ct * 16 + l15] = f2bf(c[r] + bb);
        }
    }
}

// ---------- K4: score2 = mh @ nodes^T / sqrt(D); 10*tanh + priority bias + mask; softmax -> probs ----------
__global__ void k4_final(const bf16_t* __restrict__ mh, const float* __restrict__ nodes,
                         const float* __restrict__ mask, const float* __restrict__ btab,
                         const int* __restrict__ gid, const int* __restrict__ cmp,
                         float* __restrict__ out) {
    extern __shared__ char smem[];
    float* sc   = (float*)smem;                        // 16 x NPAD f32 (64512)
    float* mlds = (float*)(smem + 16 * NPAD * 4);      // 16 x 1000 f32 (64000)
    int*   glds = (int*)(smem + 16 * NPAD * 4 + 64000);// 1000 i32 (4000)
    int pt = blockIdx.x, b = blockIdx.y;
    int tid = threadIdx.x, wid = tid >> 5, lane = tid & 31;
    int half = lane >> 4, l15 = lane & 15;

    // ---- async prefetch: mask panel (16x1000 f32) + group_ids row (1000 i32) ----
    for (int cch = tid; cch < 4000; cch += 256) {
        int row = cch / 250;
        int off = cch - row * 250;
        int p = pt * 16 + row;
        const float* g = mask + ((size_t)b * PP + (p < PP ? p : 0)) * NN + off * 4;
        async_copy_b128(mlds + row * 1000 + off * 4, g);
    }
    if (tid < 250) async_copy_b128(glds + tid * 4, gid + (size_t)b * NN + tid * 4);

    int prow = pt * 16 + l15;
    bool pv = prow < PP;
    const bf16_t* ar = mh + ((size_t)b * PP + (pv ? prow : 0)) * DD;

    // preload A (16 x 128 bf16 as 4 K-chunks) once per wave
    v16bf a0 = load_a_bf16(ar, 0,  half, pv);
    v16bf a1 = load_a_bf16(ar, 32, half, pv);
    v16bf a2 = load_a_bf16(ar, 64, half, pv);
    v16bf a3 = load_a_bf16(ar, 96, half, pv);

    for (int t = wid; t < 63; t += 8) {
        int nb = t * 16;
        int n = nb + l15;
        bool nvld = n < NN;
        const float* nrow = nodes + ((size_t)b * NN + (nvld ? n : 0)) * DD;
        if (t + 8 < 63) {
            __builtin_prefetch(nodes + ((size_t)b * NN + (t + 8) * 16 + l15) * DD, 0, 1);
        }
        v16bf b0 = load_b_f32(nrow + 0  + half * 16, nvld);
        v16bf b1 = load_b_f32(nrow + 32 + half * 16, nvld);
        v16bf b2 = load_b_f32(nrow + 64 + half * 16, nvld);
        v16bf b3 = load_b_f32(nrow + 96 + half * 16, nvld);
        v8f c = {};
        c = wmma_bf16(a0, b0, c);
        c = wmma_bf16(a1, b1, c);
        c = wmma_bf16(a2, b2, c);
        c = wmma_bf16(a3, b3, c);
        if (nvld) {
#pragma unroll
            for (int r = 0; r < 8; ++r) sc[(r + 8 * half) * NPAD + n] = c[r];
        }
    }
    wait_async0();
    __syncthreads();

    const float inv_sqrt_d = 0.08838834764831845f;  // 1/sqrt(128)
    for (int rr = 0; rr < 2; ++rr) {
        int row = wid * 2 + rr;
        int p = pt * 16 + row;
        if (p >= PP) continue;  // wave-uniform
        const float* mrow = mlds + row * 1000;
        int cm = cmp[(size_t)b * PP + p];
        float mx = -3.0e38f;
        for (int n = lane; n < NN; n += 32) {
            float s = sc[row * NPAD + n] * inv_sqrt_d;
            float cl = 10.0f * tanhf(s);
            float m = mrow[n];
            float bia = 0.0f;
            if (m == 0.0f) {  // mask is 0.0 or -inf => finite test
                int d = glds[n] - cm;
                d = d < 0 ? 0 : (d > 4 ? 4 : d);
                bia = btab[d];
            }
            float v = cl + bia + m;
            sc[row * NPAD + n] = v;
            mx = fmaxf(mx, v);
        }
        mx = wave_max(mx);
        float sum = 0.f;
        for (int n = lane; n < NN; n += 32) {
            float e = __expf(sc[row * NPAD + n] - mx);
            sc[row * NPAD + n] = e;
            sum += e;
        }
        sum = wave_sum(sum);
        float inv = 1.0f / sum;
        float* orow = out + ((size_t)b * PP + p) * NN;
        for (int n = lane; n < NN; n += 32) orow[n] = sc[row * NPAD + n] * inv;
    }
}

// ---------- launch ----------
extern "C" void kernel_launch(void* const* d_in, const int* in_sizes, int n_in,
                              void* d_out, int out_size, void* d_ws, size_t ws_size,
                              hipStream_t stream) {
    (void)in_sizes; (void)n_in; (void)out_size; (void)ws_size;
    const float* enc_nodes = (const float*)d_in[0];
    const float* enc_q1    = (const float*)d_in[1];
    const float* enc_last  = (const float*)d_in[2];
    const float* ninf      = (const float*)d_in[3];
    const float* Wqf       = (const float*)d_in[4];
    const float* Wql       = (const float*)d_in[5];
    const float* Wk        = (const float*)d_in[6];
    const float* Wv        = (const float*)d_in[7];
    const float* Wc        = (const float*)d_in[8];
    const float* bcomb     = (const float*)d_in[9];
    const float* btab      = (const float*)d_in[10];
    const int*   gid       = (const int*)d_in[11];
    const int*   cmp       = (const int*)d_in[12];
    float* out = (float*)d_out;

    // workspace layout (bytes), all 256-aligned
    char* w = (char*)d_ws;
    bf16_t* q_bf = (bf16_t*)(w + 0);          // 64*100*128*2  = 1,638,400
    bf16_t* k_bf = (bf16_t*)(w + 1638400);    // 64*1000*128*2 = 16,384,000
    bf16_t* vT   = (bf16_t*)(w + 18022400);   // 64*128*1024*2 = 16,777,216
    bf16_t* outc = (bf16_t*)(w + 34799616);   // 1,638,400
    bf16_t* mh   = (bf16_t*)(w + 36438016);   // 1,638,400
    bf16_t* wT   = (bf16_t*)(w + 38076416);   // 5*128*128*2 = 163,840  (end: 38,240,256)

    bf16_t* WqfT = wT + 0 * 16384;
    bf16_t* WqlT = wT + 1 * 16384;
    bf16_t* WkT  = wT + 2 * 16384;
    bf16_t* WvT  = wT + 3 * 16384;
    bf16_t* WcT  = wT + 4 * 16384;

    kprep_wt<<<dim3(DD, 5), DD, 0, stream>>>(Wqf, Wql, Wk, Wv, Wc, wT);

    k0_proj_q<<<dim3(7, BB), 32, 0, stream>>>(enc_q1, enc_last, WqfT, WqlT, q_bf);
    k1_proj_kv<<<dim3(63, BB), 32, 0, stream>>>(enc_nodes, WkT, WvT, k_bf, vT);

    // 64512 (scores) + 32768 (weights) + 1024 (acc) + 64000 (mask panel) = 162,304 B
    size_t smem2 = (size_t)16 * NPAD * 4 + (size_t)16 * KPAD * 2 + 1024 + 64000;
    k2_attention<<<dim3(7, HH, BB), 256, smem2, stream>>>(q_bf, k_bf, vT, ninf, outc);

    k3_comb<<<dim3(7, BB), 32, 0, stream>>>(outc, WcT, bcomb, mh);

    // 64512 (scores) + 64000 (mask panel) + 4000 (group ids) = 132,512 B
    size_t smem4 = (size_t)16 * NPAD * 4 + 64000 + 4000;
    k4_final<<<dim3(7, BB), 256, smem4, stream>>>(mh, enc_nodes, ninf, btab, gid, cmp, out);
}
